// MultiScaleAttention_72679436583193
// MI455X (gfx1250) — compile-verified, matrix operand order checked by reading, not run on hardware
//
#include <hip/hip_runtime.h>
#include <math.h>

typedef __attribute__((ext_vector_type(16))) _Float16 v16h;
typedef __attribute__((ext_vector_type(8)))  _Float16 v8h;
typedef __attribute__((ext_vector_type(8)))  float    v8f;
typedef __attribute__((ext_vector_type(4)))  float    v4f;

#define DCONST 1024
#define LROWS  512
#define NHEADS 16
#define HDIM   64

#define V8F_ZERO ((v8f){0.f,0.f,0.f,0.f,0.f,0.f,0.f,0.f})

__device__ __forceinline__ v8f wmma16x16x32(v16h a, v16h b, v8f c) {
  // 8 args: (neg_a, A, neg_b, B, c_mod, C, reuse_a, reuse_b)
  return __builtin_amdgcn_wmma_f32_16x16x32_f16(false, a, false, b, (short)0, c,
                                                false, false);
}

// Fragment layout (CDNA5 ISA 7.12.2, wave32):
//   A (16x32 f16): lane l holds row m=l&15; half h -> K(h)=16*(h/8)+(h%8)+8*(l>>4)
//   B (32x16 f16): lane l holds col n=l&15; same per-lane K pattern.
// => per lane: two contiguous 8-half (16B) chunks at (k0+8*hi) and (k0+8*hi+16).
__device__ __forceinline__ v16h load_frag_h(const _Float16* p) {
  v8h lo = *(const v8h*)(p);
  v8h hh = *(const v8h*)(p + 16);
  v16h r;
#pragma unroll
  for (int i = 0; i < 8; ++i) { r[i] = lo[i]; r[i + 8] = hh[i]; }
  return r;
}

// Same fragment but source is f32 in global memory, converted in-register.
__device__ __forceinline__ v16h load_frag_f32cvt(const float* p) {
  v4f a0 = *(const v4f*)(p);
  v4f a1 = *(const v4f*)(p + 4);
  v4f a2 = *(const v4f*)(p + 16);
  v4f a3 = *(const v4f*)(p + 20);
  v16h r;
#pragma unroll
  for (int i = 0; i < 4; ++i) {
    r[i]      = (_Float16)a0[i];
    r[i + 4]  = (_Float16)a1[i];
    r[i + 8]  = (_Float16)a2[i];
    r[i + 12] = (_Float16)a3[i];
  }
  return r;
}

// C = A(MxK, f16 rowmajor) * W(NxK, f32 rowmajor)^T + bias
// MODE 0: out f32 rowmajor [M][N]; 1: out f16 rowmajor; 2: out f16 transposed [N][M]
template <int MODE>
__global__ __launch_bounds__(256) void gemm_xwt(
    const _Float16* __restrict__ A, const float* __restrict__ W,
    const float* __restrict__ bias, void* __restrict__ outp,
    int M, int N, int Kd)
{
  const int lane = threadIdx.x & 31;
  const int l15  = lane & 15;
  const int hi8  = (lane >> 4) * 8;
  const int wave = (int)((blockIdx.x * blockDim.x + threadIdx.x) >> 5);
  const int mtiles = M >> 4;
  const int ntiles = N >> 6;
  if (wave >= mtiles * ntiles) return;
  const int m0 = (wave % mtiles) << 4;
  const int n0 = (wave / mtiles) << 6;

  v8f acc[4];
#pragma unroll
  for (int nb = 0; nb < 4; ++nb) acc[nb] = V8F_ZERO;

  const _Float16* ap = A + (size_t)(m0 + l15) * Kd + hi8;
  const float* wp[4];
#pragma unroll
  for (int nb = 0; nb < 4; ++nb)
    wp[nb] = W + (size_t)(n0 + nb * 16 + l15) * Kd + hi8;

  for (int k0 = 0; k0 < Kd; k0 += 32) {
    v16h a = load_frag_h(ap + k0);
#pragma unroll
    for (int nb = 0; nb < 4; ++nb) {
      __builtin_prefetch(wp[nb] + k0 + 128, 0, 1);  // global_prefetch_b8
      v16h b = load_frag_f32cvt(wp[nb] + k0);
      acc[nb] = wmma16x16x32(a, b, acc[nb]);
    }
  }

#pragma unroll
  for (int nb = 0; nb < 4; ++nb) {
    const int n = n0 + nb * 16 + l15;
    const float bv = bias[n];
    if (MODE == 0) {
      float* O = (float*)outp;
#pragma unroll
      for (int j = 0; j < 8; ++j)
        O[(size_t)(m0 + j + hi8) * N + n] = acc[nb][j] + bv;
    } else if (MODE == 1) {
      _Float16* O = (_Float16*)outp;
#pragma unroll
      for (int j = 0; j < 8; ++j)
        O[(size_t)(m0 + j + hi8) * N + n] = (_Float16)(acc[nb][j] + bv);
    } else {
      _Float16* O = (_Float16*)outp;
      v8h pack;
#pragma unroll
      for (int j = 0; j < 8; ++j) pack[j] = (_Float16)(acc[nb][j] + bv);
      *(v8h*)(O + (size_t)n * M + m0 + hi8) = pack;  // 16B store, m-contiguous
    }
  }
}

// Flash-style causal attention; one wave handles (scale, head, 16-query tile).
// Q,K: [3][512][1024] f16 rowmajor. Vt: [3][1024][512] f16 (transposed).
// Out: [3][512][1024] f16 rowmajor. scores = (Q.K^T) * sqrt(64) = *8 (reference bug kept).
__global__ __launch_bounds__(32) void attn_kernel(
    const _Float16* __restrict__ Q, const _Float16* __restrict__ Kmat,
    const _Float16* __restrict__ Vt, _Float16* __restrict__ Out)
{
  __shared__ __align__(16) _Float16 P[16 * 32];
  const int lane = threadIdx.x & 31;
  const int l15  = lane & 15;
  const int hi8  = (lane >> 4) * 8;
  const int qt = blockIdx.x;
  const int h  = blockIdx.y;
  const int sc = blockIdx.z;
  const size_t plane = (size_t)sc * LROWS * DCONST;
  const int m0 = qt * 16;
  const int c0 = h * HDIM;

  const _Float16* Qs = Q + plane;
  const _Float16* Ks = Kmat + plane;
  const _Float16* Vs = Vt + plane;

  const _Float16* qp = Qs + (size_t)(m0 + l15) * DCONST + c0 + hi8;
  const v16h aq0 = load_frag_h(qp);
  const v16h aq1 = load_frag_h(qp + 32);

  float mrun[8], lrun[8];
  v8f oacc[4];
#pragma unroll
  for (int j = 0; j < 8; ++j) { mrun[j] = -__builtin_inff(); lrun[j] = 0.0f; }
#pragma unroll
  for (int nb = 0; nb < 4; ++nb) oacc[nb] = V8F_ZERO;

  const int nt = (qt >> 1) + 1;  // ceil((m0+16)/32) key-32 tiles (causal)
  for (int t = 0; t < nt; ++t) {
    const int kb = t * 32;
    const _Float16* kp0 = Ks + (size_t)(kb + l15) * DCONST + c0 + hi8;
    const _Float16* kp1 = kp0 + (size_t)16 * DCONST;
    v8f s0 = V8F_ZERO, s1 = V8F_ZERO;
    s0 = wmma16x16x32(aq0, load_frag_h(kp0), s0);
    s0 = wmma16x16x32(aq1, load_frag_h(kp0 + 32), s0);
    s1 = wmma16x16x32(aq0, load_frag_h(kp1), s1);
    s1 = wmma16x16x32(aq1, load_frag_h(kp1 + 32), s1);

    const bool last = (t == nt - 1);  // only the last tile can touch the diagonal
#pragma unroll
    for (int j = 0; j < 8; ++j) {
      const int qrow = m0 + j + hi8;  // C-layout: VGPR j, lane half -> row
      float v0 = s0[j] * 8.0f;
      float v1 = s1[j] * 8.0f;
      if (last) {
        if (kb + l15 > qrow)      v0 = -__builtin_inff();
        if (kb + 16 + l15 > qrow) v1 = -__builtin_inff();
      }
      float mx = fmaxf(v0, v1);
#pragma unroll
      for (int d = 1; d < 16; d <<= 1) mx = fmaxf(mx, __shfl_xor(mx, d, 32));
      const float mnew = fmaxf(mrun[j], mx);
      const float scl = __expf(mrun[j] - mnew);
      const float e0 = __expf(v0 - mnew);
      const float e1 = __expf(v1 - mnew);
      float rs = e0 + e1;
#pragma unroll
      for (int d = 1; d < 16; d <<= 1) rs += __shfl_xor(rs, d, 32);
      lrun[j] = lrun[j] * scl + rs;
      mrun[j] = mnew;
#pragma unroll
      for (int nb = 0; nb < 4; ++nb) oacc[nb][j] *= scl;
      P[(j + hi8) * 32 + l15]      = (_Float16)e0;
      P[(j + hi8) * 32 + 16 + l15] = (_Float16)e1;
    }
    __syncthreads();
    const v16h apr = load_frag_h(&P[l15 * 32 + hi8]);  // re-stripe P as A-fragment
#pragma unroll
    for (int nb = 0; nb < 4; ++nb) {
      const _Float16* vp = Vs + (size_t)(c0 + nb * 16 + l15) * LROWS + kb + hi8;
      oacc[nb] = wmma16x16x32(apr, load_frag_h(vp), oacc[nb]);
    }
    __syncthreads();
  }

#pragma unroll
  for (int j = 0; j < 8; ++j) {
    const float inv = 1.0f / (lrun[j] + 1e-6f);  // reference: e / (sum(e) + 1e-6)
#pragma unroll
    for (int nb = 0; nb < 4; ++nb) oacc[nb][j] *= inv;
  }
  _Float16* Op = Out + plane;
#pragma unroll
  for (int nb = 0; nb < 4; ++nb)
#pragma unroll
    for (int j = 0; j < 8; ++j)
      Op[(size_t)(m0 + j + hi8) * DCONST + c0 + nb * 16 + l15] = (_Float16)oacc[nb][j];
}

__device__ __forceinline__ float block_sum_256(float v) {
  __shared__ float sm[8];
#pragma unroll
  for (int d = 16; d >= 1; d >>= 1) v += __shfl_xor(v, d, 32);
  if ((threadIdx.x & 31) == 0) sm[threadIdx.x >> 5] = v;
  __syncthreads();
  float r = 0.0f;
#pragma unroll
  for (int i = 0; i < 8; ++i) r += sm[i];
  __syncthreads();
  return r;
}

// LayerNorm over 1024, write f16 into the [512][3072] concat buffer.
__global__ __launch_bounds__(256) void ln_concat_kernel(
    const float* __restrict__ X, const float* __restrict__ g,
    const float* __restrict__ b, _Float16* __restrict__ cat)
{
  const int idx = blockIdx.x;  // s*512 + row
  const int sc = idx >> 9;
  const int row = idx & 511;
  const float* xr = X + (size_t)idx * DCONST;
  float v[4];
  float sum = 0.0f;
#pragma unroll
  for (int k = 0; k < 4; ++k) { v[k] = xr[threadIdx.x + 256 * k]; sum += v[k]; }
  sum = block_sum_256(sum);
  const float mu = sum * (1.0f / DCONST);
  float sq = 0.0f;
#pragma unroll
  for (int k = 0; k < 4; ++k) { const float d = v[k] - mu; sq += d * d; }
  sq = block_sum_256(sq);
  const float rstd = rsqrtf(sq * (1.0f / DCONST) + 1e-5f);
#pragma unroll
  for (int k = 0; k < 4; ++k) {
    const int c = threadIdx.x + 256 * k;
    cat[(size_t)row * (3 * DCONST) + sc * DCONST + c] =
        (_Float16)((v[k] - mu) * rstd * g[sc * DCONST + c] + b[sc * DCONST + c]);
  }
}

// Final LayerNorm, f32 output.
__global__ __launch_bounds__(256) void ln_out_kernel(
    const float* __restrict__ X, const float* __restrict__ g,
    const float* __restrict__ b, float* __restrict__ out)
{
  const int row = blockIdx.x;
  const float* xr = X + (size_t)row * DCONST;
  float v[4];
  float sum = 0.0f;
#pragma unroll
  for (int k = 0; k < 4; ++k) { v[k] = xr[threadIdx.x + 256 * k]; sum += v[k]; }
  sum = block_sum_256(sum);
  const float mu = sum * (1.0f / DCONST);
  float sq = 0.0f;
#pragma unroll
  for (int k = 0; k < 4; ++k) { const float d = v[k] - mu; sq += d * d; }
  sq = block_sum_256(sq);
  const float rstd = rsqrtf(sq * (1.0f / DCONST) + 1e-5f);
#pragma unroll
  for (int k = 0; k < 4; ++k) {
    const int c = threadIdx.x + 256 * k;
    out[(size_t)row * DCONST + c] = (v[k] - mu) * rstd * g[c] + b[c];
  }
}

__global__ __launch_bounds__(256) void cvt_f16_kernel(
    const float* __restrict__ in, _Float16* __restrict__ out, int n)
{
  const int i = blockIdx.x * 256 + threadIdx.x;
  if (i < n) out[i] = (_Float16)in[i];
}

extern "C" void kernel_launch(void* const* d_in, const int* in_sizes, int n_in,
                              void* d_out, int out_size, void* d_ws, size_t ws_size,
                              hipStream_t stream) {
  (void)in_sizes; (void)n_in; (void)out_size; (void)ws_size;
  const float* x    = (const float*)d_in[0];
  const float* Wq   = (const float*)d_in[1];
  const float* bq   = (const float*)d_in[2];
  const float* Wk   = (const float*)d_in[3];
  const float* bk   = (const float*)d_in[4];
  const float* Wv   = (const float*)d_in[5];
  const float* bv   = (const float*)d_in[6];
  const float* Wo   = (const float*)d_in[7];
  const float* bo   = (const float*)d_in[8];
  const float* lng  = (const float*)d_in[9];
  const float* lnb  = (const float*)d_in[10];
  const float* Wout = (const float*)d_in[11];
  const float* bout = (const float*)d_in[12];
  const float* ln2g = (const float*)d_in[13];
  const float* ln2b = (const float*)d_in[14];

  // Workspace layout (~24 MB total)
  char* p = (char*)d_ws;
  _Float16* xh   = (_Float16*)p; p += (size_t)LROWS * DCONST * 2;        // 1 MB
  _Float16* Qb   = (_Float16*)p; p += (size_t)3 * LROWS * DCONST * 2;    // 3 MB
  _Float16* Kb   = (_Float16*)p; p += (size_t)3 * LROWS * DCONST * 2;    // 3 MB
  _Float16* Vtb  = (_Float16*)p; p += (size_t)3 * LROWS * DCONST * 2;    // 3 MB  [D][L]
  _Float16* attn = (_Float16*)p; p += (size_t)3 * LROWS * DCONST * 2;    // 3 MB
  float*    pre1 = (float*)p;    p += (size_t)3 * LROWS * DCONST * 4;    // 6 MB
  _Float16* cat  = (_Float16*)p; p += (size_t)LROWS * 3 * DCONST * 2;    // 3 MB
  float*    pre2 = (float*)p;    p += (size_t)LROWS * DCONST * 4;        // 2 MB

  cvt_f16_kernel<<<(LROWS * DCONST) / 256, 256, 0, stream>>>(x, xh, LROWS * DCONST);

  const int gemmBlocks = ((LROWS / 16) * (DCONST / 64) + 7) / 8;  // 512 tiles / 8 waves
  for (int s = 0; s < 3; ++s) {
    const size_t wOff = (size_t)s * DCONST * DCONST;
    const size_t bOff = (size_t)s * DCONST;
    const size_t aOff = (size_t)s * LROWS * DCONST;
    gemm_xwt<1><<<gemmBlocks, 256, 0, stream>>>(xh, Wq + wOff, bq + bOff, Qb + aOff,
                                                LROWS, DCONST, DCONST);
    gemm_xwt<1><<<gemmBlocks, 256, 0, stream>>>(xh, Wk + wOff, bk + bOff, Kb + aOff,
                                                LROWS, DCONST, DCONST);
    gemm_xwt<2><<<gemmBlocks, 256, 0, stream>>>(xh, Wv + wOff, bv + bOff, Vtb + aOff,
                                                LROWS, DCONST, DCONST);
  }

  attn_kernel<<<dim3(LROWS / 16, NHEADS, 3), 32, 0, stream>>>(Qb, Kb, Vtb, attn);

  for (int s = 0; s < 3; ++s) {
    const size_t wOff = (size_t)s * DCONST * DCONST;
    const size_t bOff = (size_t)s * DCONST;
    const size_t aOff = (size_t)s * LROWS * DCONST;
    gemm_xwt<0><<<gemmBlocks, 256, 0, stream>>>(attn + aOff, Wo + wOff, bo + bOff,
                                                pre1 + aOff, LROWS, DCONST, DCONST);
  }

  ln_concat_kernel<<<3 * LROWS, 256, 0, stream>>>(pre1, lng, lnb, cat);

  gemm_xwt<0><<<gemmBlocks, 256, 0, stream>>>(cat, Wout, bout, pre2,
                                              LROWS, DCONST, 3 * DCONST);

  ln_out_kernel<<<LROWS, 256, 0, stream>>>(pre2, ln2g, ln2b, (float*)d_out);
}
